// DualAttentionDecoder_34909494182076
// MI455X (gfx1250) — matrix operand order repaired and checked by reading, not compiled, not run on hardware
//
#include <hip/hip_runtime.h>
#include <math.h>

// ---------------------------------------------------------------------------
// DualAttentionDecoder step for MI455X (gfx1250, wave32, WMMA)
// B=32, S=2048, D=A=CTX=H=OUT=1024, ADD=256, DEC_IN=2304
// ---------------------------------------------------------------------------

typedef __attribute__((ext_vector_type(16))) __bf16 v16bf;
typedef __attribute__((ext_vector_type(8)))  float  v8f;

union FragBf { v16bf v; unsigned int u[8]; };

__device__ __forceinline__ unsigned short f2bf(float f) {
  unsigned int x = __float_as_uint(f);
  x += 0x7FFFu + ((x >> 16) & 1u);      // round to nearest even
  return (unsigned short)(x >> 16);
}
__device__ __forceinline__ unsigned int pack2bf(float a, float b) {
  return (unsigned int)f2bf(a) | ((unsigned int)f2bf(b) << 16);
}
__device__ __forceinline__ float sigm(float x) { return 1.0f / (1.0f + expf(-x)); }

// ---------------------------------------------------------------------------
// fp32 -> bf16 conversion (for Wh)
// ---------------------------------------------------------------------------
__global__ void cvt_bf16_kernel(const float* __restrict__ src,
                                unsigned short* __restrict__ dst, int n) {
  int i = blockIdx.x * 256 + threadIdx.x;
  if (i < n) dst[i] = f2bf(src[i]);
}

// ---------------------------------------------------------------------------
// context transpose: ctxT[c][b] = ctx[b][c]   (CTX=1024, B=32)
// ---------------------------------------------------------------------------
__global__ void ctxT_kernel(const float* __restrict__ ctx, float* __restrict__ ctxT) {
  int i = blockIdx.x * 256 + threadIdx.x;       // 32768
  int c = i >> 5, b = i & 31;
  ctxT[c * 32 + b] = ctx[b * 1024 + c];
}

// ---------------------------------------------------------------------------
// ca[b][a] = sum_c context[b][c] * Wc[a][c]      (32 x 1024)
// ---------------------------------------------------------------------------
__global__ __launch_bounds__(256) void ca_kernel(const float* __restrict__ Wc,
                                                 const float* __restrict__ ctxT,
                                                 float* __restrict__ ca) {
  __shared__ float xs[64 * 32];
  int b = threadIdx.x & 31, wj = threadIdx.x >> 5;
  int a = blockIdx.x * 8 + wj;
  float acc = 0.0f;
  for (int k0 = 0; k0 < 1024; k0 += 64) {
    __syncthreads();
    for (int i = threadIdx.x; i < 64 * 32; i += 256) xs[i] = ctxT[k0 * 32 + i];
    __syncthreads();
    const float* wr = Wc + (size_t)a * 1024 + k0;
#pragma unroll 8
    for (int kk = 0; kk < 64; ++kk) acc += wr[kk] * xs[kk * 32 + b];
  }
  ca[b * 1024 + a] = acc;
}

// ---------------------------------------------------------------------------
// Attention scores (pre-softmax), fused GEMM + tanh + v-dot, bf16 WMMA.
// Block: 64 rows of (b,s); full-K (1024) history tile in LDS as bf16;
// Wh tiles double-buffered (prefetch t+1 overlaps WMMA on t, 1 barrier/tile).
// grid = (1024, 2): y=0 -> scores_ours (theirs seq), y=1 -> scores_theirs.
// ---------------------------------------------------------------------------
#define TM 64
#define KROW 1032            // padded bf16/row: 2064B = 129*16 -> b128-aligned, bank+4/row

__global__ __launch_bounds__(256) void score_kernel(
    const float* __restrict__ hist0, const int* __restrict__ len0,
    const float* __restrict__ hist1, const int* __restrict__ len1,
    const unsigned short* __restrict__ whbf,   // [A=1024][D=1024] bf16
    const float* __restrict__ ca,              // [B][A]
    const float* __restrict__ vvec,            // [A]
    float* __restrict__ scoreRaw)              // [2][B][S]
{
  __shared__ unsigned short histT[TM][KROW];   // 132,096 B
  __shared__ unsigned short whT[2][64][72];    //  18,432 B (144B rows: 16B-aligned, bank+36/row)
  __shared__ float caRow[1024];                //   4,096 B
  __shared__ float vRow[1024];                 //   4,096 B
  __shared__ float rowSum[TM];                 //     256 B   => 158,976 B total

  const int which = blockIdx.y;
  const float* hist = which ? hist1 : hist0;
  const int* lens = which ? len1 : len0;
  const int rowBase = blockIdx.x * TM;         // row in [0, B*S)
  const int b = rowBase >> 11;                 // S=2048
  const int s0 = rowBase & 2047;
  const int tid = threadIdx.x;

  // Load 64x1024 fp32 history tile -> bf16 LDS (coalesced float4 rows)
  for (int i = tid; i < TM * 256; i += 256) {
    int r = i >> 8, c4 = i & 255;
    float4 f = ((const float4*)hist)[(size_t)(rowBase + r) * 256 + c4];
    int c = c4 * 4;
    *(unsigned int*)&histT[r][c]     = pack2bf(f.x, f.y);
    *(unsigned int*)&histT[r][c + 2] = pack2bf(f.z, f.w);
  }
  for (int i = tid; i < 1024; i += 256) {
    caRow[i] = ca[b * 1024 + i];
    vRow[i]  = vvec[i];
  }
  if (tid < TM) rowSum[tid] = 0.0f;

  const int wave  = tid >> 5;                  // 8 waves (wave32)
  const int lane  = tid & 31;
  const int mSub  = wave & 3;                  // M subtile 0..3
  const int nPair = wave >> 2;                 // N subtile pair 0..1
  const int mRow  = mSub * 16 + (lane & 15);   // A-frag: lane -> row
  const int rOff  = (lane < 16) ? 0 : 8;       // D-frag row half
  const int khA   = (lane < 16) ? 0 : 16;      // A-frag K-half offset
  const int khB   = (lane < 16) ? 0 : 16;      // B-frag K-half offset
  const int nc0   = nPair * 32 + (lane & 15);  // B/D-frag: lane -> column

  // Wh tile prefetch state: tile t = (nChunk<<4)|kChunk, 64x64 bf16 = 512 uint4
  const uint4* whv = (const uint4*)whbf;       // Wh row = 128 uint4
  const int e0 = tid, e1 = tid + 256;
  uint4 r0, r1;
  {
    r0 = whv[(size_t)(e0 >> 3) * 128 + (e0 & 7)];
    r1 = whv[(size_t)(e1 >> 3) * 128 + (e1 & 7)];
    *(uint4*)&whT[0][e0 >> 3][(e0 & 7) * 8] = r0;
    *(uint4*)&whT[0][e1 >> 3][(e1 & 7) * 8] = r1;
  }

  float part[8];
#pragma unroll
  for (int r = 0; r < 8; ++r) part[r] = 0.0f;
  v8f acc0 = {0.f, 0.f, 0.f, 0.f, 0.f, 0.f, 0.f, 0.f};
  v8f acc1 = acc0;

  for (int t = 0; t < 256; ++t) {
    const int buf = t & 1;
    __syncthreads();                           // tile t visible; buf^1 free for overwrite
    if (t + 1 < 256) {                         // prefetch tile t+1 (overlaps compute)
      const int nC = (t + 1) >> 4, kC = (t + 1) & 15;
      const size_t base = (size_t)nC * 64 * 128 + (size_t)kC * 8;
      r0 = whv[base + (size_t)(e0 >> 3) * 128 + (e0 & 7)];
      r1 = whv[base + (size_t)(e1 >> 3) * 128 + (e1 & 7)];
    }
#pragma unroll
    for (int kStep = 0; kStep < 2; ++kStep) {
      const int k0 = (t & 15) * 64 + kStep * 32;
      FragBf fa, fb0, fb1;
#pragma unroll
      for (int vv = 0; vv < 8; ++vv) {         // A 16x32: 32B contiguous run per lane
        int kk = k0 + khA + 2 * vv;
        fa.u[vv] = *(const unsigned int*)&histT[mRow][kk];
      }
      const int kb = kStep * 32 + khB;         // B 32x16: 32B contiguous run per lane
#pragma unroll
      for (int vv = 0; vv < 8; ++vv) {
        fb0.u[vv] = *(const unsigned int*)&whT[buf][nc0][kb + 2 * vv];
        fb1.u[vv] = *(const unsigned int*)&whT[buf][nc0 + 16][kb + 2 * vv];
      }
      acc0 = __builtin_amdgcn_wmma_f32_16x16x32_bf16(false, fa.v, false, fb0.v,
                                                     (short)0, acc0, false, false);
      acc1 = __builtin_amdgcn_wmma_f32_16x16x32_bf16(false, fa.v, false, fb1.v,
                                                     (short)0, acc1, false, false);
    }
    if (t + 1 < 256) {                         // stash prefetched tile in other buffer
      *(uint4*)&whT[buf ^ 1][e0 >> 3][(e0 & 7) * 8] = r0;
      *(uint4*)&whT[buf ^ 1][e1 >> 3][(e1 & 7) * 8] = r1;
    }
    if ((t & 15) == 15) {                      // end of nChunk: fuse tanh(ha+ca)*v
      const int n0 = (t >> 4) * 64 + nc0;
#pragma unroll
      for (int r = 0; r < 8; ++r) {
        part[r] += tanhf(acc0[r] + caRow[n0])      * vRow[n0];
        part[r] += tanhf(acc1[r] + caRow[n0 + 16]) * vRow[n0 + 16];
      }
      acc0 = (v8f){0.f, 0.f, 0.f, 0.f, 0.f, 0.f, 0.f, 0.f};
      acc1 = acc0;
    }
  }
#pragma unroll
  for (int r = 0; r < 8; ++r)
    atomicAdd(&rowSum[mSub * 16 + r + rOff], part[r]);   // ds_add_f32
  __syncthreads();

  if (tid < TM) {
    int s = s0 + tid;
    float val = (s >= lens[b]) ? -INFINITY : rowSum[tid];
    scoreRaw[((size_t)which * 32 + b) * 2048 + s] = val;
  }
}

// ---------------------------------------------------------------------------
// Masked softmax over s per (which, b); writes ws copy + d_out scores.
// ---------------------------------------------------------------------------
__global__ __launch_bounds__(256) void softmax_kernel(float* __restrict__ scoreRaw,
                                                      const int* __restrict__ len0,
                                                      const int* __restrict__ len1,
                                                      float* __restrict__ outScores) {
  __shared__ float red[256];
  const int blk = blockIdx.x;                  // [0,64): which*32 + b
  const int which = blk >> 5;
  const int b = blk & 31;
  float* row = scoreRaw + (size_t)blk * 2048;
  const int len = (which ? len1 : len0)[b];
  const int tid = threadIdx.x;

  float m = -INFINITY;
  for (int s = tid; s < len; s += 256) m = fmaxf(m, row[s]);
  red[tid] = m; __syncthreads();
  for (int off = 128; off > 0; off >>= 1) {
    if (tid < off) red[tid] = fmaxf(red[tid], red[tid + off]);
    __syncthreads();
  }
  m = red[0]; __syncthreads();

  float sum = 0.0f;
  for (int s = tid; s < len; s += 256) sum += expf(row[s] - m);
  red[tid] = sum; __syncthreads();
  for (int off = 128; off > 0; off >>= 1) {
    if (tid < off) red[tid] += red[tid + off];
    __syncthreads();
  }
  float inv = 1.0f / red[0];

  float* dst = outScores + (size_t)blk * 2048;
  for (int s = tid; s < 2048; s += 256) {
    float v = (s < len) ? expf(row[s] - m) * inv : 0.0f;
    row[s] = v;
    dst[s] = v;
  }
}

// ---------------------------------------------------------------------------
// att context: xT[which*1024+d][b] = sum_s score[which][b][s]*hist[b][s][d]
// ---------------------------------------------------------------------------
__global__ __launch_bounds__(256) void att_kernel(const float* __restrict__ hist0,
                                                  const int* __restrict__ len0,
                                                  const float* __restrict__ hist1,
                                                  const int* __restrict__ len1,
                                                  const float* __restrict__ score,
                                                  float* __restrict__ xT) {
  const int which = blockIdx.z;
  const int b = blockIdx.y;
  const int d = blockIdx.x * 256 + threadIdx.x;
  const float* hist = which ? hist1 : hist0;
  const int len = (which ? len1 : len0)[b];
  const float* sc = score + ((size_t)which * 32 + b) * 2048;
  const float* hrow = hist + (size_t)b * 2048 * 1024 + d;
  float acc = 0.0f;
  int s = 0;
  for (; s + 4 <= len; s += 4) {
    acc += sc[s]     * hrow[(size_t)s * 1024];
    acc += sc[s + 1] * hrow[(size_t)(s + 1) * 1024];
    acc += sc[s + 2] * hrow[(size_t)(s + 2) * 1024];
    acc += sc[s + 3] * hrow[(size_t)(s + 3) * 1024];
  }
  for (; s < len; ++s) acc += sc[s] * hrow[(size_t)s * 1024];
  xT[(size_t)(which * 1024 + d) * 32 + b] = acc;
}

// xT tail: rows 2048..2303 = additional_decoder_input
__global__ void addl_kernel(const float* __restrict__ addl, float* __restrict__ xT) {
  int i = blockIdx.x * 256 + threadIdx.x;      // 8192
  int k = i >> 5, b = i & 31;
  xT[(size_t)(2048 + k) * 32 + b] = addl[b * 256 + k];
}

// gather h0[hidden_idx], h1[hidden_idx] into transposed [K][32]
__global__ void gather_kernel(const float* __restrict__ h0, const float* __restrict__ h1,
                              const int* __restrict__ idx,
                              float* __restrict__ h0T, float* __restrict__ h1T) {
  int i = blockIdx.x * 256 + threadIdx.x;      // 32768
  int k = i >> 5, b = i & 31;
  h0T[k * 32 + b] = h0[(size_t)idx[b] * 1024 + k];
  h1T[k * 32 + b] = h1[(size_t)idx[b] * 1024 + k];
}

// ---------------------------------------------------------------------------
// gatesT[j][b] = bih[j]+bhh[j] + sum_k Wih[j][k]*xT[k][b] + sum_k Whh[j][k]*hT[k][b]
// ---------------------------------------------------------------------------
__global__ __launch_bounds__(256) void gates_kernel(const float* __restrict__ xT, int K1,
                                                    const float* __restrict__ hT, int K2,
                                                    const float* __restrict__ Wih,
                                                    const float* __restrict__ Whh,
                                                    const float* __restrict__ bih,
                                                    const float* __restrict__ bhh,
                                                    float* __restrict__ gatesT) {
  __shared__ float xs[64 * 32];
  int b = threadIdx.x & 31, wj = threadIdx.x >> 5;
  int j = blockIdx.x * 8 + wj;
  float acc = bih[j] + bhh[j];
  for (int k0 = 0; k0 < K1; k0 += 64) {
    __syncthreads();
    for (int i = threadIdx.x; i < 64 * 32; i += 256) xs[i] = xT[(size_t)k0 * 32 + i];
    __syncthreads();
    const float* wr = Wih + (size_t)j * K1 + k0;
#pragma unroll 8
    for (int kk = 0; kk < 64; ++kk) acc += wr[kk] * xs[kk * 32 + b];
  }
  for (int k0 = 0; k0 < K2; k0 += 64) {
    __syncthreads();
    for (int i = threadIdx.x; i < 64 * 32; i += 256) xs[i] = hT[(size_t)k0 * 32 + i];
    __syncthreads();
    const float* wr = Whh + (size_t)j * K2 + k0;
#pragma unroll 8
    for (int kk = 0; kk < 64; ++kk) acc += wr[kk] * xs[kk * 32 + b];
  }
  gatesT[j * 32 + b] = acc;
}

// LSTM pointwise: i,f,g,o -> c_new/h_new; scatter to d_out state slots
__global__ void lstm_point_kernel(const float* __restrict__ gatesT,
                                  const float* __restrict__ cIn,
                                  const int* __restrict__ idx,
                                  float* __restrict__ hSlot, float* __restrict__ cSlot,
                                  float* __restrict__ hOutT) {
  int i = blockIdx.x * 256 + threadIdx.x;      // 32768
  int h = i >> 5, b = i & 31;
  float ig = sigm(gatesT[h * 32 + b]);
  float fg = sigm(gatesT[(1024 + h) * 32 + b]);
  float gg = tanhf(gatesT[(2048 + h) * 32 + b]);
  float og = sigm(gatesT[(3072 + h) * 32 + b]);
  int row = idx[b];
  float c = cIn[(size_t)row * 1024 + h];
  float cn = fg * c + ig * gg;
  float hn = og * tanhf(cn);
  cSlot[(size_t)row * 1024 + h] = cn;
  hSlot[(size_t)row * 1024 + h] = hn;
  hOutT[h * 32 + b] = hn;
}

// out[b][o] = tanh(bout[o] + sum_k Wout[o][k]*h1T[k][b])
__global__ __launch_bounds__(256) void out_kernel(const float* __restrict__ Wout,
                                                  const float* __restrict__ bout,
                                                  const float* __restrict__ hT,
                                                  float* __restrict__ out) {
  __shared__ float xs[64 * 32];
  int b = threadIdx.x & 31, wj = threadIdx.x >> 5;
  int o = blockIdx.x * 8 + wj;
  float acc = bout[o];
  for (int k0 = 0; k0 < 1024; k0 += 64) {
    __syncthreads();
    for (int i = threadIdx.x; i < 64 * 32; i += 256) xs[i] = hT[k0 * 32 + i];
    __syncthreads();
    const float* wr = Wout + (size_t)o * 1024 + k0;
#pragma unroll 8
    for (int kk = 0; kk < 64; ++kk) acc += wr[kk] * xs[kk * 32 + b];
  }
  out[b * 1024 + o] = tanhf(acc);
}

// ---------------------------------------------------------------------------
extern "C" void kernel_launch(void* const* d_in, const int* in_sizes, int n_in,
                              void* d_out, int out_size, void* d_ws, size_t ws_size,
                              hipStream_t stream) {
  (void)in_sizes; (void)n_in; (void)out_size; (void)ws_size;

  const float* seqOurs   = (const float*)d_in[0];
  const int*   lenOurs   = (const int*)d_in[1];
  const float* seqTheirs = (const float*)d_in[2];
  const int*   lenTheirs = (const int*)d_in[3];
  const float* ctx       = (const float*)d_in[4];
  const float* addl      = (const float*)d_in[5];
  const float* h0        = (const float*)d_in[6];
  const float* c0        = (const float*)d_in[7];
  const float* h1        = (const float*)d_in[8];
  const float* c1        = (const float*)d_in[9];
  const int*   hidx      = (const int*)d_in[10];
  const float* attWh     = (const float*)d_in[11];
  const float* attWc     = (const float*)d_in[12];
  const float* attV      = (const float*)d_in[13];
  const float* Wih0      = (const float*)d_in[14];
  const float* Whh0      = (const float*)d_in[15];
  const float* bih0      = (const float*)d_in[16];
  const float* bhh0      = (const float*)d_in[17];
  const float* Wih1      = (const float*)d_in[18];
  const float* Whh1      = (const float*)d_in[19];
  const float* bih1      = (const float*)d_in[20];
  const float* bhh1      = (const float*)d_in[21];
  const float* Wout      = (const float*)d_in[22];
  const float* bout      = (const float*)d_in[23];

  // d_out: out | h0_new | c0_new | h1_new | c1_new | scores_ours | scores_theirs
  float* outP = (float*)d_out;
  float* h0n = outP + 32768;
  float* c0n = outP + 65536;
  float* h1n = outP + 98304;
  float* c1n = outP + 131072;
  float* scoresOut = outP + 163840;

  char* ws = (char*)d_ws;
  unsigned short* whBf = (unsigned short*)(ws);          // 2 MB  bf16 Wh
  float* ca       = (float*)(ws + 2097152);              // 128 KB
  float* scoreRaw = (float*)(ws + 2228224);              // 512 KB
  float* ctxT     = (float*)(ws + 2752512);              // 128 KB
  float* xT       = (float*)(ws + 2883584);              // 288 KB
  float* h0selT   = (float*)(ws + 3178496);              // 128 KB
  float* h1selT   = (float*)(ws + 3309568);              // 128 KB
  float* gatesT   = (float*)(ws + 3440640);              // 512 KB
  float* h0outT   = (float*)(ws + 3964928);              // 128 KB
  float* h1outT   = (float*)(ws + 4096000);              // 128 KB -> ~4.03 MB total

  cvt_bf16_kernel<<<4096, 256, 0, stream>>>(attWh, whBf, 1024 * 1024);
  ctxT_kernel<<<128, 256, 0, stream>>>(ctx, ctxT);
  ca_kernel<<<128, 256, 0, stream>>>(attWc, ctxT, ca);

  // which=0: scores_ours from theirs-seq; which=1: scores_theirs from ours-seq
  score_kernel<<<dim3(1024, 2), 256, 0, stream>>>(seqTheirs, lenTheirs, seqOurs, lenOurs,
                                                  whBf, ca, attV, scoreRaw);
  softmax_kernel<<<64, 256, 0, stream>>>(scoreRaw, lenTheirs, lenOurs, scoresOut);
  att_kernel<<<dim3(4, 32, 2), 256, 0, stream>>>(seqTheirs, lenTheirs, seqOurs, lenOurs,
                                                 scoreRaw, xT);
  addl_kernel<<<32, 256, 0, stream>>>(addl, xT);
  gather_kernel<<<128, 256, 0, stream>>>(h0, h1, hidx, h0selT, h1selT);

  hipMemcpyAsync(h0n, h0, 32768 * sizeof(float), hipMemcpyDeviceToDevice, stream);
  hipMemcpyAsync(c0n, c0, 32768 * sizeof(float), hipMemcpyDeviceToDevice, stream);
  hipMemcpyAsync(h1n, h1, 32768 * sizeof(float), hipMemcpyDeviceToDevice, stream);
  hipMemcpyAsync(c1n, c1, 32768 * sizeof(float), hipMemcpyDeviceToDevice, stream);

  gates_kernel<<<512, 256, 0, stream>>>(xT, 2304, h0selT, 1024, Wih0, Whh0, bih0, bhh0, gatesT);
  lstm_point_kernel<<<128, 256, 0, stream>>>(gatesT, c0, hidx, h0n, c0n, h0outT);
  gates_kernel<<<512, 256, 0, stream>>>(h0outT, 1024, h1selT, 1024, Wih1, Whh1, bih1, bhh1, gatesT);
  lstm_point_kernel<<<128, 256, 0, stream>>>(gatesT, c1, hidx, h1n, c1n, h1outT);
  out_kernel<<<128, 256, 0, stream>>>(Wout, bout, h1outT, outP);
}
